// HogLayer3D_28518582845877
// MI455X (gfx1250) — compile-verified, matrix operand order checked by reading, not run on hardware
//
#include <hip/hip_runtime.h>

typedef __attribute__((ext_vector_type(16))) _Float16 v16h;
typedef __attribute__((ext_vector_type(8)))  float    v8f;

#define HDIM 128
#define CELL 8
#define CPA  16          // cells per axis (128/8)
#define TILE 10          // 8 + 2 halo
#define BINS 20
#define BLK_O 15         // CPA - 2 + 1

// Dodecahedron vertices (unnormalized; all have norm sqrt(3), so uniform
// scaling does not change the argmax bin).
__device__ __constant__ float VTAB[20][3] = {
  { 1.f, 1.f, 1.f}, {-1.f,-1.f,-1.f}, { 1.f, 1.f,-1.f}, {-1.f,-1.f, 1.f},
  { 1.f,-1.f, 1.f}, {-1.f, 1.f,-1.f}, { 1.f,-1.f,-1.f}, {-1.f, 1.f, 1.f},
  {0.f, 0.6180339887f, 1.6180339887f}, {0.f,-0.6180339887f,-1.6180339887f},
  {0.f, 0.6180339887f,-1.6180339887f}, {0.f,-0.6180339887f, 1.6180339887f},
  { 1.6180339887f, 0.f, 0.6180339887f}, {-1.6180339887f, 0.f,-0.6180339887f},
  { 1.6180339887f, 0.f,-0.6180339887f}, {-1.6180339887f, 0.f, 0.6180339887f},
  { 0.6180339887f, 1.6180339887f, 0.f}, {-0.6180339887f,-1.6180339887f, 0.f},
  { 0.6180339887f,-1.6180339887f, 0.f}, {-0.6180339887f, 1.6180339887f, 0.f}
};

__device__ __forceinline__ int reflect_idx(int g) {
  g = (g < 0) ? -g : g;
  g = (g >= HDIM) ? (2 * HDIM - 2 - g) : g;
  return g;
}

// One workgroup per 8x8x8 cell: Sobel gradients from an LDS tile, WMMA
// projection onto 20 directions, branchless cross-lane argmax, LDS histogram.
__global__ __launch_bounds__(512) void hog_cell_kernel(
    const float* __restrict__ x, float* __restrict__ cellHist) {
  __shared__ float tile[TILE * TILE * TILE];
  __shared__ float hist[BINS];

  const int tid = threadIdx.x;
  const int cid = blockIdx.x;
  const int cd = cid & 15;
  const int cw = (cid >> 4) & 15;
  const int ch = (cid >> 8) & 15;
  const int bb = cid >> 12;

  if (tid < BINS) hist[tid] = 0.0f;

  // Load 10x10x10 reflect-padded tile.
  const long base = (long)bb * ((long)HDIM * HDIM * HDIM);
  for (int i = tid; i < TILE * TILE * TILE; i += 512) {
    int td = i % TILE;
    int tw = (i / TILE) % TILE;
    int th = i / (TILE * TILE);
    int gH = reflect_idx(ch * CELL + th - 1);
    int gW = reflect_idx(cw * CELL + tw - 1);
    int gD = reflect_idx(cd * CELL + td - 1);
    tile[i] = x[base + ((long)gH * HDIM + gW) * HDIM + gD];
  }
  __syncthreads();

  // Per-voxel 3x3x3 Sobel (cross-correlation, as XLA conv does).
  const int ld = tid & 7, lw = (tid >> 3) & 7, lh = tid >> 6;
  const float wgt[3] = {1.f, 2.f, 1.f};
  const float dif[3] = {-1.f, 0.f, 1.f};
  float gh = 0.f, gw = 0.f, gd = 0.f;
#pragma unroll
  for (int a = 0; a < 3; ++a)
#pragma unroll
    for (int b = 0; b < 3; ++b)
#pragma unroll
      for (int c = 0; c < 3; ++c) {
        float t = tile[(lh + a) * (TILE * TILE) + (lw + b) * TILE + (ld + c)];
        gh += wgt[b] * wgt[c] * dif[a] * t;
        gw += wgt[a] * wgt[c] * dif[b] * t;
        gd += wgt[a] * wgt[b] * dif[c] * t;
      }
  const float nrm = sqrtf(gh * gh + gw * gw + gd * gd);

  const int lane = tid & 31;
  const int l16 = lane & 15;

  // B matrices: column N = l16 holds K=0..2 = V[:,N] (lanes 0-15 only).
  v16h B0 = {};
  v16h B1 = {};
  if (lane < 16) {
    B0[0] = (_Float16)VTAB[l16][0];
    B0[1] = (_Float16)VTAB[l16][1];
    B0[2] = (_Float16)VTAB[l16][2];
    if (l16 < 4) {
      B1[0] = (_Float16)VTAB[16 + l16][0];
      B1[1] = (_Float16)VTAB[16 + l16][1];
      B1[2] = (_Float16)VTAB[16 + l16][2];
    }
  }

#pragma unroll
  for (int bat = 0; bat < 2; ++bat) {
    // Batch voxels = wave lanes [bat*16, bat*16+16). Route gradients so that
    // lanes 0-15 hold rows M=0..15 with K=0..2 = (gh,gw,gd).
    float g0 = bat ? __shfl_xor(gh, 16) : gh;
    float g1 = bat ? __shfl_xor(gw, 16) : gw;
    float g2 = bat ? __shfl_xor(gd, 16) : gd;
    v16h A = {};
    if (lane < 16) {
      A[0] = (_Float16)g0;
      A[1] = (_Float16)g1;
      A[2] = (_Float16)g2;
    }
    v8f c0 = {};
    v8f c1 = {};
    c0 = __builtin_amdgcn_wmma_f32_16x16x32_f16(false, A, false, B0,
                                                (short)0, c0, false, false);
    c1 = __builtin_amdgcn_wmma_f32_16x16x32_f16(false, A, false, B1,
                                                (short)0, c1, false, false);

    // D layout: VGPR r = phase[M=r][N=lane] (lanes 0-15), phase[M=8+r] (16-31).
#pragma unroll
    for (int r = 0; r < 8; ++r) {
      // Merge bins 0..15 (c0) with bins 16..19 (c1); strict > keeps the
      // lower bin index on ties (argmax picks first max).
      float va = c0[r];
      float vb = (l16 < 4) ? c1[r] : -3.0e38f;
      bool tb = vb > va;
      float v = tb ? vb : va;
      int idx = tb ? (16 + l16) : l16;

      // Branchless butterfly argmax within each 16-lane half.
#pragma unroll
      for (int m = 8; m >= 1; m >>= 1) {
        float ov = __shfl_xor(v, m);
        int   oi = __shfl_xor(idx, m);
        bool take = (ov > v) | ((ov == v) & (oi < idx));
        v = take ? ov : v;
        idx = take ? oi : idx;
      }

      const int vA = bat * 16 + r;      // result now in lanes 0-15
      const int vB = bat * 16 + 8 + r;  // result now in lanes 16-31
      float nA = __shfl(nrm, vA);       // uniform src lane -> readlane
      float nB = __shfl(nrm, vB);
      float contrib = (lane < 16) ? nA : nB;
      if (l16 == 0) atomicAdd(&hist[idx], contrib);  // lanes 0 and 16
    }
  }
  __syncthreads();
  if (tid < BINS) cellHist[(long)cid * BINS + tid] = hist[tid];
}

// One block per output position: gather 8 cells x 20 bins in torch order
// (bin, di, dj, dk), L2-normalize with max(n, 1e-5).
__global__ __launch_bounds__(160) void hog_block_kernel(
    const float* __restrict__ cellHist, float* __restrict__ out) {
  __shared__ float ssum;
  const int t = threadIdx.x;
  const int bid = blockIdx.x;
  const int k = bid % BLK_O;
  const int j = (bid / BLK_O) % BLK_O;
  const int i = (bid / (BLK_O * BLK_O)) % BLK_O;
  const int bb = bid / (BLK_O * BLK_O * BLK_O);

  const int bin = t >> 3;
  const int off = t & 7;
  const int di = (off >> 2) & 1, dj = (off >> 1) & 1, dk = off & 1;
  const int cid = ((bb * CPA + (i + di)) * CPA + (j + dj)) * CPA + (k + dk);
  const float val = cellHist[(long)cid * BINS + bin];

  if (t == 0) ssum = 0.0f;
  __syncthreads();
  float sq = val * val;
#pragma unroll
  for (int m = 16; m >= 1; m >>= 1) sq += __shfl_xor(sq, m);
  if ((t & 31) == 0) atomicAdd(&ssum, sq);
  __syncthreads();

  const float n = sqrtf(ssum);
  out[(long)bid * (BINS * 8) + t] = val / fmaxf(n, 1e-5f);
}

extern "C" void kernel_launch(void* const* d_in, const int* in_sizes, int n_in,
                              void* d_out, int out_size, void* d_ws, size_t ws_size,
                              hipStream_t stream) {
  const float* x = (const float*)d_in[0];
  float* out = (float*)d_out;
  float* cellHist = (float*)d_ws;  // Bn * 4096 cells * 20 bins * 4B

  const int Bn = in_sizes[0] / (HDIM * HDIM * HDIM);
  hog_cell_kernel<<<Bn * CPA * CPA * CPA, 512, 0, stream>>>(x, cellHist);
  hog_block_kernel<<<Bn * BLK_O * BLK_O * BLK_O, 160, 0, stream>>>(cellHist, out);
}
